// GrConv_5626407157789
// MI455X (gfx1250) — matrix operand order repaired and checked by reading, not compiled, not run on hardware
//
#include <hip/hip_runtime.h>

#define HH 2048   // hidden size; M rows = 512*2*16 = 16384 ; right row = left row + 32

typedef __attribute__((ext_vector_type(16))) __bf16 v16bf;
typedef __attribute__((ext_vector_type(8)))  __bf16 v8bf;
typedef __attribute__((ext_vector_type(4)))  __bf16 v4bf;
typedef __attribute__((ext_vector_type(8)))  float  v8f;
typedef __attribute__((ext_vector_type(4)))  float  f4;

// ---------------- workspace layout (bf16 elements) ----------------
#define X_ELEMS   (16416u * 2048u)       // 33,619,968  (all 513*2*16 rows)
#define W_ELEMS   (2048u * 2048u)        //  4,194,304
#define G_ELEMS   (3u * 2048u * 2048u)   // 12,582,912
#define OFS_X     0u
#define OFS_WL    (OFS_X  + X_ELEMS)
#define OFS_WR    (OFS_WL + W_ELEMS)
#define OFS_GL    (OFS_WR + W_ELEMS)
#define OFS_GR    (OFS_GL + G_ELEMS)
#define TOT_ELEMS (OFS_GR + G_ELEMS)     // 67,174,400 elems = ~128.1 MiB

// =================== pre-pass: fp32 -> bf16 ===================
__global__ __launch_bounds__(256) void cvt_f32_to_bf16(
    const float* __restrict__ src, __bf16* __restrict__ dst, unsigned n) {
  unsigned i = (blockIdx.x * 256u + threadIdx.x) * 4u;   // n is a multiple of 4
  if (i < n) {
    f4 v = *(const f4*)(src + i);
    v4bf o;
    o[0] = (__bf16)v[0]; o[1] = (__bf16)v[1]; o[2] = (__bf16)v[2]; o[3] = (__bf16)v[3];
    *(v4bf*)(dst + i) = o;
  }
}

// =================== shared helpers ===================
__device__ __forceinline__ v8f wmma_bf16(v16bf a, v16bf b, v8f c) {
  return __builtin_amdgcn_wmma_f32_16x16x32_bf16(
      /*neg_a=*/false, a, /*neg_b=*/false, b,
      /*c_mod=*/(short)0, c, /*reuse_a=*/false, /*reuse_b=*/false);
}

__device__ __forceinline__ void epilogue(const float* __restrict__ x,
                                         float* __restrict__ out,
                                         int mbase, int hi, int col,
                                         v8f accC, v8f g0v, v8f g1v, v8f g2v) {
#pragma unroll
  for (int r = 0; r < 8; ++r) {
    const int    mrow = mbase + hi * 8 + r;          // C/D layout: M = vgpr + 8*(lane>=16)
    const size_t off  = (size_t)mrow * HH + col;
    const float lv = x[off];
    const float rv = x[off + (size_t)32 * HH];
    const float cN = 1.0f / (1.0f + __expf(-accC[r]));
    const float a0 = g0v[r], a1 = g1v[r], a2 = g2v[r];
    const float mx = fmaxf(a0, fmaxf(a1, a2));
    const float e0 = __expf(a0 - mx);
    const float e1 = __expf(a1 - mx);
    const float e2 = __expf(a2 - mx);
    const float inv = 1.0f / (e0 + e1 + e2);
    __builtin_nontemporal_store((lv * e0 + rv * e1 + cN * e2) * inv, out + off);
  }
}

// =================== fast path: bf16 operands from workspace ===================
// A fragment (16x32): lane = row (lane&15), K base = 16*(lane>>4);
//   elems 0..15 <- p[0..15]   (bf16 units)
__device__ __forceinline__ v16bf load_a_bf(const __bf16* __restrict__ p) {
  v8bf lo = *(const v8bf*)(p);
  v8bf hi = *(const v8bf*)(p + 8);
  return __builtin_shufflevector(lo, hi, 0,1,2,3,4,5,6,7,8,9,10,11,12,13,14,15);
}
// B fragment (32x16): lane = column (lane&15), K base = 16*(lane>>4); 16 contiguous
__device__ __forceinline__ v16bf load_b_bf(const __bf16* __restrict__ p) {
  return *(const v16bf*)(p);
}

__global__ __launch_bounds__(256) void grconv_bf16_kernel(
    const float*  __restrict__ x,    // fp32 (epilogue only)
    const __bf16* __restrict__ xb,
    const __bf16* __restrict__ WL, const __bf16* __restrict__ WR,
    const __bf16* __restrict__ GL, const __bf16* __restrict__ GR,
    float* __restrict__ out) {
  const int lane = threadIdx.x & 31;
  const int wave = threadIdx.x >> 5;

  const int ntile = blockIdx.y * 8 + wave;   // 0..127
  const int n0    = ntile * 16;
  const int m0    = blockIdx.x * 32;         // two 16-row M-tiles per wave

  const int row = lane & 15;
  const int hi  = lane >> 4;
  const int col = n0 + row;

  const __bf16* pL0 = xb + (size_t)(m0 + row) * HH + (size_t)hi * 16;
  const __bf16* pL1 = pL0 + (size_t)16 * HH;

  const size_t koff = (size_t)hi * 16;
  const __bf16* pWL  = WL + (size_t)col * HH + koff;
  const __bf16* pWR  = WR + (size_t)col * HH + koff;
  const __bf16* pGL0 = GL + (size_t)col * HH + koff;
  const __bf16* pGL1 = GL + (size_t)(HH + col) * HH + koff;
  const __bf16* pGL2 = GL + (size_t)(2 * HH + col) * HH + koff;
  const __bf16* pGR0 = GR + (size_t)col * HH + koff;
  const __bf16* pGR1 = GR + (size_t)(HH + col) * HH + koff;
  const __bf16* pGR2 = GR + (size_t)(2 * HH + col) * HH + koff;

  v8f accC0 = {}, accG00 = {}, accG10 = {}, accG20 = {};
  v8f accC1 = {}, accG01 = {}, accG11 = {}, accG21 = {};

  for (int k = 0; k < HH; k += 32) {
    const v16bf aL0 = load_a_bf(pL0 + k);
    const v16bf aR0 = load_a_bf(pL0 + (size_t)32 * HH + k);
    const v16bf aL1 = load_a_bf(pL1 + k);
    const v16bf aR1 = load_a_bf(pL1 + (size_t)32 * HH + k);

    v16bf b;
    b = load_b_bf(pWL + k);
    accC0  = wmma_bf16(aL0, b, accC0);  accC1  = wmma_bf16(aL1, b, accC1);
    b = load_b_bf(pWR + k);
    accC0  = wmma_bf16(aR0, b, accC0);  accC1  = wmma_bf16(aR1, b, accC1);

    b = load_b_bf(pGL0 + k);
    accG00 = wmma_bf16(aL0, b, accG00); accG01 = wmma_bf16(aL1, b, accG01);
    b = load_b_bf(pGR0 + k);
    accG00 = wmma_bf16(aR0, b, accG00); accG01 = wmma_bf16(aR1, b, accG01);

    b = load_b_bf(pGL1 + k);
    accG10 = wmma_bf16(aL0, b, accG10); accG11 = wmma_bf16(aL1, b, accG11);
    b = load_b_bf(pGR1 + k);
    accG10 = wmma_bf16(aR0, b, accG10); accG11 = wmma_bf16(aR1, b, accG11);

    b = load_b_bf(pGL2 + k);
    accG20 = wmma_bf16(aL0, b, accG20); accG21 = wmma_bf16(aL1, b, accG21);
    b = load_b_bf(pGR2 + k);
    accG20 = wmma_bf16(aR0, b, accG20); accG21 = wmma_bf16(aR1, b, accG21);
  }

  epilogue(x, out, m0,      hi, col, accC0, accG00, accG10, accG20);
  epilogue(x, out, m0 + 16, hi, col, accC1, accG01, accG11, accG21);
}

// =================== fallback: inline fp32->bf16 (no workspace) ===================
__device__ __forceinline__ v16bf pack16(f4 a, f4 b, f4 c, f4 d) {
  v16bf v;
  v[0]  = (__bf16)a[0]; v[1]  = (__bf16)a[1]; v[2]  = (__bf16)a[2]; v[3]  = (__bf16)a[3];
  v[4]  = (__bf16)b[0]; v[5]  = (__bf16)b[1]; v[6]  = (__bf16)b[2]; v[7]  = (__bf16)b[3];
  v[8]  = (__bf16)c[0]; v[9]  = (__bf16)c[1]; v[10] = (__bf16)c[2]; v[11] = (__bf16)c[3];
  v[12] = (__bf16)d[0]; v[13] = (__bf16)d[1]; v[14] = (__bf16)d[2]; v[15] = (__bf16)d[3];
  return v;
}
__device__ __forceinline__ v16bf load_a_f32(const float* __restrict__ p) {
  return pack16(*(const f4*)(p),      *(const f4*)(p + 4),
                *(const f4*)(p + 8),  *(const f4*)(p + 12));
}
__device__ __forceinline__ v16bf load_b_f32(const float* __restrict__ p) {
  return pack16(*(const f4*)(p),     *(const f4*)(p + 4),
                *(const f4*)(p + 8), *(const f4*)(p + 12));
}

__global__ __launch_bounds__(256) void grconv_f32_kernel(
    const float* __restrict__ x,  const float* __restrict__ WL,
    const float* __restrict__ WR, const float* __restrict__ GL,
    const float* __restrict__ GR, float* __restrict__ out) {
  const int lane = threadIdx.x & 31;
  const int wave = threadIdx.x >> 5;
  const int ntile = blockIdx.y * 8 + wave;
  const int n0    = ntile * 16;
  const int m0    = blockIdx.x * 32;
  const int row = lane & 15;
  const int hi  = lane >> 4;
  const int col = n0 + row;

  const float* pL0 = x + (size_t)(m0 + row) * HH + (size_t)hi * 16;
  const float* pL1 = pL0 + (size_t)16 * HH;
  const size_t koff = (size_t)hi * 16;
  const float* pWL  = WL + (size_t)col * HH + koff;
  const float* pWR  = WR + (size_t)col * HH + koff;
  const float* pGL0 = GL + (size_t)col * HH + koff;
  const float* pGL1 = GL + (size_t)(HH + col) * HH + koff;
  const float* pGL2 = GL + (size_t)(2 * HH + col) * HH + koff;
  const float* pGR0 = GR + (size_t)col * HH + koff;
  const float* pGR1 = GR + (size_t)(HH + col) * HH + koff;
  const float* pGR2 = GR + (size_t)(2 * HH + col) * HH + koff;

  v8f accC0 = {}, accG00 = {}, accG10 = {}, accG20 = {};
  v8f accC1 = {}, accG01 = {}, accG11 = {}, accG21 = {};

  for (int k = 0; k < HH; k += 32) {
    const v16bf aL0 = load_a_f32(pL0 + k);
    const v16bf aR0 = load_a_f32(pL0 + (size_t)32 * HH + k);
    const v16bf aL1 = load_a_f32(pL1 + k);
    const v16bf aR1 = load_a_f32(pL1 + (size_t)32 * HH + k);

    v16bf b;
    b = load_b_f32(pWL + k);
    accC0  = wmma_bf16(aL0, b, accC0);  accC1  = wmma_bf16(aL1, b, accC1);
    b = load_b_f32(pWR + k);
    accC0  = wmma_bf16(aR0, b, accC0);  accC1  = wmma_bf16(aR1, b, accC1);
    b = load_b_f32(pGL0 + k);
    accG00 = wmma_bf16(aL0, b, accG00); accG01 = wmma_bf16(aL1, b, accG01);
    b = load_b_f32(pGR0 + k);
    accG00 = wmma_bf16(aR0, b, accG00); accG01 = wmma_bf16(aR1, b, accG01);
    b = load_b_f32(pGL1 + k);
    accG10 = wmma_bf16(aL0, b, accG10); accG11 = wmma_bf16(aL1, b, accG11);
    b = load_b_f32(pGR1 + k);
    accG10 = wmma_bf16(aR0, b, accG10); accG11 = wmma_bf16(aR1, b, accG11);
    b = load_b_f32(pGL2 + k);
    accG20 = wmma_bf16(aL0, b, accG20); accG21 = wmma_bf16(aL1, b, accG21);
    b = load_b_f32(pGR2 + k);
    accG20 = wmma_bf16(aR0, b, accG20); accG21 = wmma_bf16(aR1, b, accG21);
  }

  epilogue(x, out, m0,      hi, col, accC0, accG00, accG10, accG20);
  epilogue(x, out, m0 + 16, hi, col, accC1, accG01, accG11, accG21);
}

// =================== host ===================
extern "C" void kernel_launch(void* const* d_in, const int* in_sizes, int n_in,
                              void* d_out, int out_size, void* d_ws, size_t ws_size,
                              hipStream_t stream) {
  (void)in_sizes; (void)n_in; (void)out_size;
  const float* x  = (const float*)d_in[0];
  const float* WL = (const float*)d_in[1];
  const float* WR = (const float*)d_in[2];
  const float* GL = (const float*)d_in[3];
  const float* GR = (const float*)d_in[4];
  float* out = (float*)d_out;

  dim3 grid(512, 16, 1);    // 16384 rows / 32 per block ; 128 ntiles / 8 waves
  dim3 block(256, 1, 1);

  const size_t need = (size_t)TOT_ELEMS * sizeof(__bf16);
  if (d_ws != nullptr && ws_size >= need) {
    __bf16* wsb = (__bf16*)d_ws;
    // fp32 -> bf16 pre-pass (memory bound)
    cvt_f32_to_bf16<<<(X_ELEMS / 4 + 255) / 256, 256, 0, stream>>>(x,  wsb + OFS_X,  X_ELEMS);
    cvt_f32_to_bf16<<<(W_ELEMS / 4 + 255) / 256, 256, 0, stream>>>(WL, wsb + OFS_WL, W_ELEMS);
    cvt_f32_to_bf16<<<(W_ELEMS / 4 + 255) / 256, 256, 0, stream>>>(WR, wsb + OFS_WR, W_ELEMS);
    cvt_f32_to_bf16<<<(G_ELEMS / 4 + 255) / 256, 256, 0, stream>>>(GL, wsb + OFS_GL, G_ELEMS);
    cvt_f32_to_bf16<<<(G_ELEMS / 4 + 255) / 256, 256, 0, stream>>>(GR, wsb + OFS_GR, G_ELEMS);
    grconv_bf16_kernel<<<grid, block, 0, stream>>>(
        x, wsb + OFS_X, wsb + OFS_WL, wsb + OFS_WR, wsb + OFS_GL, wsb + OFS_GR, out);
  } else {
    grconv_f32_kernel<<<grid, block, 0, stream>>>(x, WL, WR, GL, GR, out);
  }
}
